// DiscoveryEngineModel_71184787964500
// MI455X (gfx1250) — matrix operand order repaired and checked by reading, not compiled, not run on hardware
//
#include <hip/hip_runtime.h>
#include <hip/hip_bf16.h>
#include <math.h>

typedef __attribute__((ext_vector_type(16))) _Float16 v16h;
typedef __attribute__((ext_vector_type(8)))  float    v8f;

#define WAVES 4
#define TPB   (WAVES * 32)

// SiLU via hardware reciprocal (v_rcp_f32) instead of IEEE division:
// avoids the v_div_scale/v_div_fmas Newton-Raphson chain (~12 VALU ops/element).
__device__ __forceinline__ float silu(float v) {
  return v * __builtin_amdgcn_rcpf(1.0f + __expf(-v));
}

__device__ __forceinline__ v8f wmma_f16(v16h a, v16h b, v8f c) {
  // D = A(16x32 f16) * B(32x16 f16) + C(16x16 f32)
  return __builtin_amdgcn_wmma_f32_16x16x32_f16(false, a, false, b, (short)0, c, false, false);
}

// Load B tile [32x16] (K-chunk kc, N-tile nt) from LDS weights stored transposed [64][kstride] f16.
// B layout (wave32, 16-bit): lanes 0-15 hold K = {0..7, 16..23} (N = lane), lanes 16-31 hold K = {8..15, 24..31}.
__device__ __forceinline__ v16h load_btile(const _Float16* w, int kstride, int kc, int nt, int lane) {
  int n  = nt * 16 + (lane & 15);
  int kb = kc * 32 + ((lane & 16) ? 8 : 0);
  const _Float16* p = w + n * kstride + kb;
  v16h b;
#pragma unroll
  for (int i = 0; i < 8; ++i) { b[i] = p[i]; b[8 + i] = p[16 + i]; }
  return b;
}

// Load A tile [16x32] (K-chunk kc) from activation LDS [16][64] row-major f16.
// A layout (16-bit): lanes 0-15: M=lane, K = {0..7, 16..23}; lanes 16-31: M=lane-16, K = {8..15, 24..31}.
__device__ __forceinline__ v16h load_atile(const _Float16* act, int kc, int lane) {
  int m  = lane & 15;
  int hb = (lane & 16) ? 8 : 0;
  const _Float16* row = act + m * 64 + kc * 32 + hb;
  v16h a;
#pragma unroll
  for (int j = 0; j < 8; ++j) { a[j] = row[j]; a[8 + j] = row[16 + j]; }
  return a;
}

// C init = broadcast bias along rows; D/C layout: col = (lane&15) within N-tile, same for all 8 VGPRs.
__device__ __forceinline__ v8f bias_c(const float* lb, int nt, int lane) {
  float bv = lb[nt * 16 + (lane & 15)];
  v8f c;
#pragma unroll
  for (int r = 0; r < 8; ++r) c[r] = bv;
  return c;
}

// ---------------------------------------------------------------------------
// Edge kernel: fused phi_e (10->64->64->64) + phi_v (10->64->1) + scatter-add.
// One wave processes 16 edges per tile (grid-stride over tiles).
// ---------------------------------------------------------------------------
__global__ __launch_bounds__(TPB) void egnn_edge(
    const float* __restrict__ x, const float* __restrict__ pos, const float* __restrict__ vel,
    const int* __restrict__ ei,  // [2,E] : row0 = src, row1 = dst
    const float* __restrict__ pe_w1, const float* __restrict__ pe_b1,
    const float* __restrict__ pe_w2, const float* __restrict__ pe_b2,
    const float* __restrict__ pe_w3, const float* __restrict__ pe_b3,
    const float* __restrict__ pv_w1, const float* __restrict__ pv_b1,
    const float* __restrict__ pv_w2, const float* __restrict__ pv_b2,
    float* __restrict__ agg_h, float* __restrict__ agg_v,
    int E, int ntiles)
{
  __shared__ _Float16 lw1[64 * 32];   // pe_w1 transposed, K padded 10->32
  __shared__ _Float16 lw2[64 * 64];   // pe_w2 transposed
  __shared__ _Float16 lw3[64 * 64];   // pe_w3 transposed
  __shared__ _Float16 lv1[64 * 32];   // pv_w1 transposed, K padded
  __shared__ float lb1[64], lb2[64], lb3[64], lbv1[64], lv2[64];
  __shared__ _Float16 act[WAVES][16 * 64];  // per-wave activation staging
  __shared__ int dsts[WAVES][16];           // per-wave dst indices of the 16 edges

  const int tid = threadIdx.x;
  for (int i = tid; i < 64 * 32; i += TPB) {
    int n = i >> 5, k = i & 31;
    lw1[i] = (k < 10) ? (_Float16)pe_w1[k * 64 + n] : (_Float16)0.0f;
    lv1[i] = (k < 10) ? (_Float16)pv_w1[k * 64 + n] : (_Float16)0.0f;
  }
  for (int i = tid; i < 64 * 64; i += TPB) {
    int n = i >> 6, k = i & 63;
    lw2[i] = (_Float16)pe_w2[k * 64 + n];
    lw3[i] = (_Float16)pe_w3[k * 64 + n];
  }
  for (int i = tid; i < 64; i += TPB) {
    lb1[i] = pe_b1[i]; lb2[i] = pe_b2[i]; lb3[i] = pe_b3[i];
    lbv1[i] = pv_b1[i]; lv2[i] = pv_w2[i];
  }
  __syncthreads();

  const int wave = tid >> 5, lane = tid & 31;
  const int m    = lane & 15;
  const bool lo  = (lane < 16);
  const int mrow = lo ? 0 : 8;
  _Float16* wact = act[wave];
  int*      wdst = dsts[wave];
  const float pvb2 = pv_b2[0];

  for (int t = blockIdx.x * WAVES + wave; t < ntiles; t += gridDim.x * WAVES) {
    int e = t * 16 + m;
    float emask = (e < E) ? 1.0f : 0.0f;
    if (e >= E) e = E - 1;                 // clamp; contributions masked below
    const int src = ei[e];
    const int dst = ei[E + e];

    // gathers (all L2-resident)
    float xd0 = x[dst * 4 + 0], xd1 = x[dst * 4 + 1], xd2 = x[dst * 4 + 2], xd3 = x[dst * 4 + 3];
    float xs0 = x[src * 4 + 0], xs1 = x[src * 4 + 1], xs2 = x[src * 4 + 2], xs3 = x[src * 4 + 3];
    float rpx = pos[src * 2 + 0] - pos[dst * 2 + 0];
    float rpy = pos[src * 2 + 1] - pos[dst * 2 + 1];
    float rvx = vel[src * 2 + 0] - vel[dst * 2 + 0];
    float rvy = vel[src * 2 + 1] - vel[dst * 2 + 1];
    float dsq = rpx * rpx + rpy * rpy;
    float dvr = rvx * rpx + rvy * rpy;

    // Build A = tmp[16 x 10] zero-padded to 16x32 directly in A layout.
    // lanes 0-15 hold K=0..7 (x_dst, x_src); lanes 16-31 hold K=8..9 (dsq, dvr).
    v16h a0;
#pragma unroll
    for (int i = 0; i < 16; ++i) a0[i] = (_Float16)0.0f;
    a0[0] = (_Float16)(lo ? xd0 : dsq);
    a0[1] = (_Float16)(lo ? xd1 : dvr);
    a0[2] = lo ? (_Float16)xd2 : (_Float16)0.0f;
    a0[3] = lo ? (_Float16)xd3 : (_Float16)0.0f;
    a0[4] = lo ? (_Float16)xs0 : (_Float16)0.0f;
    a0[5] = lo ? (_Float16)xs1 : (_Float16)0.0f;
    a0[6] = lo ? (_Float16)xs2 : (_Float16)0.0f;
    a0[7] = lo ? (_Float16)xs3 : (_Float16)0.0f;

    if (lo) wdst[m] = dst;   // stash dst indices for the scatter phase

    // ---- phi_e layer 1: [16x32] x [32x64] -> SiLU -> LDS ----
#pragma unroll
    for (int nt = 0; nt < 4; ++nt) {
      v8f c = bias_c(lb1, nt, lane);
      c = wmma_f16(a0, load_btile(lw1, 32, 0, nt, lane), c);
      int col = nt * 16 + m;
#pragma unroll
      for (int r = 0; r < 8; ++r) wact[(mrow + r) * 64 + col] = (_Float16)silu(c[r]);
    }

    // ---- phi_e layer 2: [16x64] x [64x64] -> SiLU -> LDS ----
    v16h h0 = load_atile(wact, 0, lane);
    v16h h1 = load_atile(wact, 1, lane);
#pragma unroll
    for (int nt = 0; nt < 4; ++nt) {
      v8f c = bias_c(lb2, nt, lane);
      c = wmma_f16(h0, load_btile(lw2, 64, 0, nt, lane), c);
      c = wmma_f16(h1, load_btile(lw2, 64, 1, nt, lane), c);
      int col = nt * 16 + m;
#pragma unroll
      for (int r = 0; r < 8; ++r) wact[(mrow + r) * 64 + col] = (_Float16)silu(c[r]);
    }

    // ---- phi_e layer 3: [16x64] x [64x64] -> scatter-add to agg_h ----
    v16h g0 = load_atile(wact, 0, lane);
    v16h g1 = load_atile(wact, 1, lane);
#pragma unroll
    for (int nt = 0; nt < 4; ++nt) {
      v8f c = bias_c(lb3, nt, lane);
      c = wmma_f16(g0, load_btile(lw3, 64, 0, nt, lane), c);
      c = wmma_f16(g1, load_btile(lw3, 64, 1, nt, lane), c);
      int col = nt * 16 + m;
#pragma unroll
      for (int r = 0; r < 8; ++r) {
        int em = mrow + r;
        if (t * 16 + em < E)
          atomicAdd(&agg_h[(long)wdst[em] * 64 + col], c[r]);
      }
    }

    // ---- phi_v layer 1: [16x32] x [32x64] -> SiLU -> LDS ----
#pragma unroll
    for (int nt = 0; nt < 4; ++nt) {
      v8f c = bias_c(lbv1, nt, lane);
      c = wmma_f16(a0, load_btile(lv1, 32, 0, nt, lane), c);
      int col = nt * 16 + m;
#pragma unroll
      for (int r = 0; r < 8; ++r) wact[(mrow + r) * 64 + col] = (_Float16)silu(c[r]);
    }

    // ---- phi_v layer 2 (64->1): split dot across lane halves, combine via shfl ----
    int hb = lo ? 0 : 32;
    float acc = 0.0f;
#pragma unroll
    for (int j = 0; j < 32; ++j) acc += (float)wact[m * 64 + hb + j] * lv2[hb + j];
    acc += __shfl_xor(acc, 16);
    float vw = acc + pvb2;
    if (lo) {
      atomicAdd(&agg_v[dst * 2 + 0], vw * rpx * emask);
      atomicAdd(&agg_v[dst * 2 + 1], vw * rpy * emask);
    }
  }
}

// ---------------------------------------------------------------------------
// Node kernel: hh = [x | agg_h | ||agg_v||]  (69 -> 64 SiLU -> 64)
// K order in padded-96 space: K0..63 = agg_h, K64..67 = x, K68 = norm.
// ---------------------------------------------------------------------------
__global__ __launch_bounds__(TPB) void egnn_node(
    const float* __restrict__ x,
    const float* __restrict__ agg_h, const float* __restrict__ agg_v,
    const float* __restrict__ ph_w1, const float* __restrict__ ph_b1,
    const float* __restrict__ ph_w2, const float* __restrict__ ph_b2,
    float* __restrict__ out, int N)
{
  __shared__ _Float16 lw1[64 * 96];
  __shared__ _Float16 lw2[64 * 64];
  __shared__ float lb1[64], lb2[64];
  __shared__ _Float16 act[WAVES][16 * 64];

  const int tid = threadIdx.x;
  for (int i = tid; i < 64 * 96; i += TPB) {
    int n = i / 96, k = i % 96;
    float w = 0.0f;
    if (k < 64)       w = ph_w1[(4 + k) * 64 + n];        // agg_h part
    else if (k < 68)  w = ph_w1[(k - 64) * 64 + n];       // x part
    else if (k == 68) w = ph_w1[68 * 64 + n];             // norm part
    lw1[i] = (_Float16)w;
  }
  for (int i = tid; i < 64 * 64; i += TPB) {
    int n = i >> 6, k = i & 63;
    lw2[i] = (_Float16)ph_w2[k * 64 + n];
  }
  for (int i = tid; i < 64; i += TPB) { lb1[i] = ph_b1[i]; lb2[i] = ph_b2[i]; }
  __syncthreads();

  const int wave = tid >> 5, lane = tid & 31;
  const int t  = blockIdx.x * WAVES + wave;
  const int nb = t * 16;
  if (nb >= N) return;
  const int m   = lane & 15;
  const bool lo = (lane < 16);
  const int mrow = lo ? 0 : 8;
  int node = nb + m; if (node >= N) node = N - 1;
  _Float16* wact = act[wave];

  // A chunks 0/1 straight from agg_h (row-contiguous per lane)
  const float* row = agg_h + (long)node * 64;
  const int hb8 = lo ? 0 : 8;
  v16h a0, a1;
#pragma unroll
  for (int j = 0; j < 8; ++j) {
    a0[j]     = (_Float16)row[hb8 + j];
    a0[8 + j] = (_Float16)row[16 + hb8 + j];
    a1[j]     = (_Float16)row[32 + hb8 + j];
    a1[8 + j] = (_Float16)row[48 + hb8 + j];
  }
  float avx = agg_v[node * 2 + 0], avy = agg_v[node * 2 + 1];
  float nrm = sqrtf(avx * avx + avy * avy);
  v16h a2;
#pragma unroll
  for (int i = 0; i < 16; ++i) a2[i] = (_Float16)0.0f;
  a2[0] = lo ? (_Float16)x[node * 4 + 0] : (_Float16)0.0f;
  a2[1] = lo ? (_Float16)x[node * 4 + 1] : (_Float16)0.0f;
  a2[2] = lo ? (_Float16)x[node * 4 + 2] : (_Float16)0.0f;
  a2[3] = lo ? (_Float16)x[node * 4 + 3] : (_Float16)0.0f;
  a2[4] = lo ? (_Float16)nrm : (_Float16)0.0f;

  // layer 1: 3 K-chunks x 4 N-tiles -> SiLU -> LDS
#pragma unroll
  for (int nt = 0; nt < 4; ++nt) {
    v8f c = bias_c(lb1, nt, lane);
    c = wmma_f16(a0, load_btile(lw1, 96, 0, nt, lane), c);
    c = wmma_f16(a1, load_btile(lw1, 96, 1, nt, lane), c);
    c = wmma_f16(a2, load_btile(lw1, 96, 2, nt, lane), c);
    int col = nt * 16 + m;
#pragma unroll
    for (int r = 0; r < 8; ++r) wact[(mrow + r) * 64 + col] = (_Float16)silu(c[r]);
  }

  // layer 2 -> coalesced global stores (nodes are contiguous in the tile)
  v16h h0 = load_atile(wact, 0, lane);
  v16h h1 = load_atile(wact, 1, lane);
#pragma unroll
  for (int nt = 0; nt < 4; ++nt) {
    v8f c = bias_c(lb2, nt, lane);
    c = wmma_f16(h0, load_btile(lw2, 64, 0, nt, lane), c);
    c = wmma_f16(h1, load_btile(lw2, 64, 1, nt, lane), c);
    int col = nt * 16 + m;
#pragma unroll
    for (int r = 0; r < 8; ++r) {
      int nr = nb + mrow + r;
      if (nr < N) out[(long)nr * 64 + col] = c[r];
    }
  }
}

__global__ void zero_f32(float* p, long n) {
  long i = (long)blockIdx.x * blockDim.x + threadIdx.x;
  long s = (long)gridDim.x * blockDim.x;
  for (; i < n; i += s) p[i] = 0.0f;
}

extern "C" void kernel_launch(void* const* d_in, const int* in_sizes, int n_in,
                              void* d_out, int out_size, void* d_ws, size_t ws_size,
                              hipStream_t stream) {
  const float* x     = (const float*)d_in[0];
  const float* pos   = (const float*)d_in[1];
  const float* vel   = (const float*)d_in[2];
  const int*   ei    = (const int*)d_in[3];
  const float* pe_w1 = (const float*)d_in[4];
  const float* pe_b1 = (const float*)d_in[5];
  const float* pe_w2 = (const float*)d_in[6];
  const float* pe_b2 = (const float*)d_in[7];
  const float* pe_w3 = (const float*)d_in[8];
  const float* pe_b3 = (const float*)d_in[9];
  const float* pv_w1 = (const float*)d_in[10];
  const float* pv_b1 = (const float*)d_in[11];
  const float* pv_w2 = (const float*)d_in[12];
  const float* pv_b2 = (const float*)d_in[13];
  const float* ph_w1 = (const float*)d_in[14];
  const float* ph_b1 = (const float*)d_in[15];
  const float* ph_w2 = (const float*)d_in[16];
  const float* ph_b2 = (const float*)d_in[17];

  const int N = in_sizes[0] / 4;
  const int E = in_sizes[3] / 2;

  float* agg_h = (float*)d_ws;                 // [N,64]
  float* agg_v = agg_h + (size_t)N * 64;       // [N,2]

  zero_f32<<<512, 256, 0, stream>>>(agg_h, (long)N * 66);

  const int etiles  = (E + 15) / 16;
  int eblocks = (etiles + WAVES - 1) / WAVES;
  if (eblocks > 2048) eblocks = 2048;          // grid-stride covers the rest
  egnn_edge<<<eblocks, TPB, 0, stream>>>(x, pos, vel, ei,
                                         pe_w1, pe_b1, pe_w2, pe_b2, pe_w3, pe_b3,
                                         pv_w1, pv_b1, pv_w2, pv_b2,
                                         agg_h, agg_v, E, etiles);

  const int ntiles  = (N + 15) / 16;
  const int nblocks = (ntiles + WAVES - 1) / WAVES;
  egnn_node<<<nblocks, TPB, 0, stream>>>(x, agg_h, agg_v,
                                         ph_w1, ph_b1, ph_w2, ph_b2,
                                         (float*)d_out, N);
}
